// Model_22677427323544
// MI455X (gfx1250) — matrix operand order repaired and checked by reading, not compiled, not run on hardware
//
#include <hip/hip_runtime.h>

// ---------------------------------------------------------------------------
// Quantum circuit expectation on MI455X (gfx1250, wave32, WMMA + async LDS).
//
// out[b] = <psi_b| U'^H Z0' U' |psi_b>  restructured as:
//   R[b, j]   : real product-state magnitudes after AngleEmbedding (f16)
//   U'[a, j]  : fixed 512x512 circuit unitary with (-i)^popc(j) phase folded
//   S = U' R^T (two real GEMMs, f16 in / f32 acc via v_wmma_f32_16x16x32_f16)
//   out[b] = sum_a sign(a) * (S_re^2 + S_im^2),  sign = +1 if a<256 else -1
//
// GEMM kernel: B operand is identical for all 4 waves of a block, so it is
// staged into LDS with global_load_async_to_lds_b128 (ASYNCcnt-tracked,
// double-buffered) and consumed with ds_load_b128; A is register
// double-buffered. This removes the load->wait0->wmma serialization and cuts
// B L2 traffic 4x.
// ---------------------------------------------------------------------------

typedef __attribute__((ext_vector_type(16))) _Float16 v16h;
typedef __attribute__((ext_vector_type(8)))  _Float16 v8h;
typedef __attribute__((ext_vector_type(8)))  float    v8f;

#define N_WIRES 9
#define DIM 512          // 2^9
#define BIT(w) (8 - (w)) // wire 0 is the MSB of the flattened index

// ---------------------- Kernel A: build U' columns -------------------------
// One block per basis column j. 512-amplitude complex state in LDS.

__device__ __forceinline__ void gate1q(float2* amp, int bp,
                                       float2 u00, float2 u01,
                                       float2 u10, float2 u11, int tid) {
  // 256 threads, one amplitude pair each.
  int lo = tid & ((1 << bp) - 1);
  int hi = tid >> bp;
  int i0 = (hi << (bp + 1)) | lo;
  int i1 = i0 | (1 << bp);
  float2 a0 = amp[i0], a1 = amp[i1];
  float2 n0, n1;
  n0.x = u00.x * a0.x - u00.y * a0.y + u01.x * a1.x - u01.y * a1.y;
  n0.y = u00.x * a0.y + u00.y * a0.x + u01.x * a1.y + u01.y * a1.x;
  n1.x = u10.x * a0.x - u10.y * a0.y + u11.x * a1.x - u11.y * a1.y;
  n1.y = u10.x * a0.y + u10.y * a0.x + u11.x * a1.y + u11.y * a1.x;
  amp[i0] = n0;
  amp[i1] = n1;
  __syncthreads();
}

__device__ __forceinline__ void gateC(float2* amp, int bpc, int bpt,
                                      float2 u00, float2 u01,
                                      float2 u10, float2 u11, int tid) {
  // Controlled 2x2: acts on the control=1 subspace (128 pairs).
  if (tid < 128) {
    int p1 = bpc < bpt ? bpc : bpt;
    int p2 = bpc < bpt ? bpt : bpc;
    int y = ((tid >> p1) << (p1 + 1)) | (tid & ((1 << p1) - 1));
    y = ((y >> p2) << (p2 + 1)) | (y & ((1 << p2) - 1));
    int i0 = y | (1 << bpc);   // control=1, target=0
    int i1 = i0 | (1 << bpt);  // control=1, target=1
    float2 a0 = amp[i0], a1 = amp[i1];
    float2 n0, n1;
    n0.x = u00.x * a0.x - u00.y * a0.y + u01.x * a1.x - u01.y * a1.y;
    n0.y = u00.x * a0.y + u00.y * a0.x + u01.x * a1.y + u01.y * a1.x;
    n1.x = u10.x * a0.x - u10.y * a0.y + u11.x * a1.x - u11.y * a1.y;
    n1.y = u10.x * a0.y + u10.y * a0.x + u11.x * a1.y + u11.y * a1.x;
    amp[i0] = n0;
    amp[i1] = n1;
  }
  __syncthreads();
}

__device__ void entangle_block(float2* amp, const float* p, int tid) {
  for (int i = 0; i < N_WIRES; i++) {
    int ip = (i + 1) % N_WIRES;
    int bi = BIT(i), bip = BIT(ip);
    float s, c;
    // RY(p[4i]) on i
    sincosf(0.5f * p[4 * i], &s, &c);
    gate1q(amp, bi, {c, 0}, {-s, 0}, {s, 0}, {c, 0}, tid);
    // RY(p[4i+1]) on ip
    sincosf(0.5f * p[4 * i + 1], &s, &c);
    gate1q(amp, bip, {c, 0}, {-s, 0}, {s, 0}, {c, 0}, tid);
    // CNOT(i, ip)
    gateC(amp, bi, bip, {0, 0}, {1, 0}, {1, 0}, {0, 0}, tid);
    // CRZ(p[4i+2])(i, ip): diag(e^{-it/2}, e^{it/2}) on control=1 subspace
    sincosf(0.5f * p[4 * i + 2], &s, &c);
    gateC(amp, bi, bip, {c, -s}, {0, 0}, {0, 0}, {c, s}, tid);
    // X on ip
    gate1q(amp, bip, {0, 0}, {1, 0}, {1, 0}, {0, 0}, tid);
    // CRX(p[4i+3])(i, ip): [[c,-is],[-is,c]] on control=1 subspace
    sincosf(0.5f * p[4 * i + 3], &s, &c);
    gateC(amp, bi, bip, {c, 0}, {0, -s}, {0, -s}, {c, 0}, tid);
  }
}

__device__ void sel_layer(float2* amp, const float* w, int r, int tid) {
  for (int i = 0; i < N_WIRES; i++) {
    int bi = BIT(i);
    float s, c;
    // Rot(phi, theta, omega) = RZ(omega) RY(theta) RZ(phi): apply RZ(phi) first
    sincosf(0.5f * w[3 * i + 0], &s, &c);
    gate1q(amp, bi, {c, -s}, {0, 0}, {0, 0}, {c, s}, tid);  // RZ(phi)
    sincosf(0.5f * w[3 * i + 1], &s, &c);
    gate1q(amp, bi, {c, 0}, {-s, 0}, {s, 0}, {c, 0}, tid);  // RY(theta)
    sincosf(0.5f * w[3 * i + 2], &s, &c);
    gate1q(amp, bi, {c, -s}, {0, 0}, {0, 0}, {c, s}, tid);  // RZ(omega)
  }
  for (int i = 0; i < N_WIRES; i++) {
    int t = (i + r) % N_WIRES;
    gateC(amp, BIT(i), BIT(t), {0, 0}, {1, 0}, {1, 0}, {0, 0}, tid);  // CNOT
  }
}

__global__ void __launch_bounds__(256)
build_U(const float* __restrict__ params, const float* __restrict__ weights,
        const float* __restrict__ params2,
        _Float16* __restrict__ Ua_re, _Float16* __restrict__ Ua_im) {
  __shared__ float2 amp[DIM];
  int tid = threadIdx.x;
  int j = blockIdx.x;  // basis column
  amp[tid] = make_float2(tid == j ? 1.f : 0.f, 0.f);
  amp[tid + 256] = make_float2((tid + 256) == j ? 1.f : 0.f, 0.f);
  __syncthreads();

  for (int l = 0; l < 3; l++) entangle_block(amp, params + l * 36, tid);
  for (int l = 0; l < 3; l++) sel_layer(amp, weights + l * 27, l + 1, tid);
  for (int l = 0; l < 5; l++) entangle_block(amp, params2 + l * 36, tid);

  // Fold the product-state phase (-i)^popc(j) into column j; store a-major
  // (Ua[a][j]) so GEMM B-fragments are contiguous per output column.
  int k = __popc(j) & 3;
  for (int a = tid; a < DIM; a += 256) {
    float2 v = amp[a];
    float re, im;
    switch (k) {
      case 0:  re =  v.x; im =  v.y; break;
      case 1:  re =  v.y; im = -v.x; break;
      case 2:  re = -v.x; im = -v.y; break;
      default: re = -v.y; im =  v.x; break;
    }
    Ua_re[a * DIM + j] = (_Float16)re;
    Ua_im[a * DIM + j] = (_Float16)im;
  }
}

// ---------------------- Kernel B: product-state magnitudes -----------------
__global__ void __launch_bounds__(256)
build_R(const float* __restrict__ adds, _Float16* __restrict__ R) {
  __shared__ float cs[N_WIRES], sn[N_WIRES];
  int b = blockIdx.x, tid = threadIdx.x;
  if (tid < N_WIRES) {
    float s, c;
    sincosf(0.5f * adds[b * N_WIRES + tid], &s, &c);
    cs[tid] = c;
    sn[tid] = s;
  }
  __syncthreads();
  for (int idx = tid; idx < DIM; idx += 256) {
    float r = 1.f;
#pragma unroll
    for (int w = 0; w < N_WIRES; w++)
      r *= ((idx >> BIT(w)) & 1) ? sn[w] : cs[w];
    R[b * DIM + idx] = (_Float16)r;
  }
}

// ---------------------- Kernel C: WMMA GEMM + epilogue ---------------------
// 4 waves / block; each wave owns one 16-row M tile; all waves share the B
// operand, which is staged into LDS via async copies (ASYNCcnt-tracked),
// double buffered across K-steps.

__device__ __forceinline__ void async_copy16(unsigned lds_addr,
                                             const _Float16* gptr) {
  // 16-byte global -> LDS async copy (per lane). Tracked by ASYNCcnt.
  asm volatile("global_load_async_to_lds_b128 %0, %1, off"
               :
               : "v"(lds_addr), "v"((unsigned long long)gptr)
               : "memory");
}

__device__ __forceinline__ void wait_async0() {
  asm volatile("s_wait_asynccnt 0x0" ::: "memory");
}

__global__ void __launch_bounds__(128)
qgemm(const _Float16* __restrict__ R, const _Float16* __restrict__ Ure,
      const _Float16* __restrict__ Uim, float* __restrict__ out) {
  // lsB[buf][frag][lane*16 halfs]; frag = 2*tile + (0:re, 1:im). 16 KB.
  __shared__ _Float16 lsB[2][8][512];

  int lane = threadIdx.x & 31;
  int wid = threadIdx.x >> 5;
  int m0 = (blockIdx.x * 4 + wid) * 16;
  int row = lane & 15;
  int loA = (lane < 16) ? 0 : 16;   // A 16x32 layout (hi lanes: K=16..31)
  int loB = (lane < 16) ? 0 : 16;   // B 32x16 layout (hi lanes: K=16..31)

  const _Float16* baseA = R + (m0 + row) * DIM + loA;

  float accOut[8];
#pragma unroll
  for (int i = 0; i < 8; i++) accOut[i] = 0.f;

  for (int ng = 0; ng < 8; ng++) {  // 8 groups x 4 N-tiles
    v8f accRe[4], accIm[4];
#pragma unroll
    for (int t = 0; t < 4; t++) { accRe[t] = {}; accIm[t] = {}; }

    // --- stage kk=0 into buffer 0: this wave stages fragments 2*wid(+1) ---
    {
      const _Float16* src0 = Ure + ((ng * 4 + wid) * 16 + row) * DIM + loB;
      const _Float16* src1 = Uim + ((ng * 4 + wid) * 16 + row) * DIM + loB;
      unsigned d0 = (unsigned)(unsigned long long)&lsB[0][2 * wid][lane * 16];
      unsigned d1 = (unsigned)(unsigned long long)&lsB[0][2 * wid + 1][lane * 16];
      async_copy16(d0, src0);
      async_copy16(d0 + 16, src0 + 8);
      async_copy16(d1, src1);
      async_copy16(d1 + 16, src1 + 8);
    }

    // --- preload A fragment for kk=0 (register double buffer) ---
    v8h alo = *(const v8h*)(baseA);
    v8h ahi = *(const v8h*)(baseA + 8);
    v16h af = __builtin_shufflevector(alo, ahi, 0, 1, 2, 3, 4, 5, 6, 7, 8, 9,
                                      10, 11, 12, 13, 14, 15);

    for (int kk = 0; kk < 16; kk++) {
      int cur = kk & 1;
      wait_async0();     // own async writes have landed in LDS
      __syncthreads();   // all waves' writes visible; prev buffer free

      // --- prefetch next K-step into the other buffer ---
      if (kk < 15) {
        int k1 = (kk + 1) * 32;
        const _Float16* src0 =
            Ure + ((ng * 4 + wid) * 16 + row) * DIM + k1 + loB;
        const _Float16* src1 =
            Uim + ((ng * 4 + wid) * 16 + row) * DIM + k1 + loB;
        unsigned d0 =
            (unsigned)(unsigned long long)&lsB[cur ^ 1][2 * wid][lane * 16];
        unsigned d1 =
            (unsigned)(unsigned long long)&lsB[cur ^ 1][2 * wid + 1][lane * 16];
        async_copy16(d0, src0);
        async_copy16(d0 + 16, src0 + 8);
        async_copy16(d1, src1);
        async_copy16(d1 + 16, src1 + 8);
      }

      // --- prefetch next A fragment ---
      v16h afn = af;
      if (kk < 15) {
        const _Float16* pA = baseA + (kk + 1) * 32;
        v8h nlo = *(const v8h*)(pA);
        v8h nhi = *(const v8h*)(pA + 8);
        afn = __builtin_shufflevector(nlo, nhi, 0, 1, 2, 3, 4, 5, 6, 7, 8, 9,
                                      10, 11, 12, 13, 14, 15);
      }

      // --- 8 WMMAs on current buffer (B from LDS, ds_load_b128 x2 each) ---
#pragma unroll
      for (int t = 0; t < 4; t++) {
        v16h br = *(const v16h*)&lsB[cur][2 * t][lane * 16];
        v16h bi = *(const v16h*)&lsB[cur][2 * t + 1][lane * 16];
        accRe[t] = __builtin_amdgcn_wmma_f32_16x16x32_f16(
            false, af, false, br, (short)0, accRe[t], false, false);
        accIm[t] = __builtin_amdgcn_wmma_f32_16x16x32_f16(
            false, af, false, bi, (short)0, accIm[t], false, false);
      }
      af = afn;
    }

#pragma unroll
    for (int t = 0; t < 4; t++) {
      float sign = ((ng * 4 + t) * 16 < 256) ? 1.f : -1.f;  // wire-0 parity
#pragma unroll
      for (int i = 0; i < 8; i++)
        accOut[i] += sign * (accRe[t][i] * accRe[t][i] +
                             accIm[t][i] * accIm[t][i]);
    }
    __syncthreads();  // done reading buf1 before next ng restages buf0
  }

  // Sum over N (lanes) within each 16-lane half: xor masks 1,2,4,8.
#pragma unroll
  for (int m = 1; m < 16; m <<= 1)
#pragma unroll
    for (int i = 0; i < 8; i++)
      accOut[i] += __shfl_xor(accOut[i], m, 32);

  // D layout: lanes 0-15 hold rows m0+0..7 (VGPR i -> M=i),
  //           lanes 16-31 hold rows m0+8..15 (VGPR i -> M=8+i).
  if ((lane & 15) == 0) {
    int base = m0 + (lane >> 4) * 8;
#pragma unroll
    for (int i = 0; i < 8; i++) out[base + i] = accOut[i];
  }
}

// ---------------------------------------------------------------------------
extern "C" void kernel_launch(void* const* d_in, const int* in_sizes, int n_in,
                              void* d_out, int out_size, void* d_ws,
                              size_t ws_size, hipStream_t stream) {
  const float* adds = (const float*)d_in[0];     // [B, 9]
  const float* params = (const float*)d_in[1];   // [108]
  const float* weights = (const float*)d_in[2];  // [3, 9, 3]
  const float* params2 = (const float*)d_in[3];  // [180]
  float* out = (float*)d_out;                    // [B]
  int B = in_sizes[0] / N_WIRES;                 // 8192

  char* ws = (char*)d_ws;
  _Float16* Ua_re = (_Float16*)ws;                            // 512 KB
  _Float16* Ua_im = (_Float16*)(ws + (size_t)DIM * DIM * 2);  // 512 KB
  _Float16* Rm = (_Float16*)(ws + (size_t)2 * DIM * DIM * 2); // B*512*2 bytes

  build_U<<<DIM, 256, 0, stream>>>(params, weights, params2, Ua_re, Ua_im);
  build_R<<<B, 256, 0, stream>>>(adds, Rm);
  qgemm<<<B / 64, 128, 0, stream>>>(Rm, Ua_re, Ua_im, out);
}
